// Estor_45595372814586
// MI455X (gfx1250) — matrix operand order repaired and checked by reading, not compiled, not run on hardware
//
#include <hip/hip_runtime.h>

// ---------------- problem constants ----------------
#define BB 8
#define SS 1024
#define DD 1024
#define FF 4096
#define HH 16
#define NSPAN 512
#define SPANL 32

typedef __attribute__((ext_vector_type(16))) __bf16 v16bf;
typedef __attribute__((ext_vector_type(8)))  float  v8f;
typedef __attribute__((ext_vector_type(4))) unsigned int u32x4;
typedef __attribute__((ext_vector_type(8))) int i32x8;
typedef __attribute__((ext_vector_type(4))) int i32x4;

__device__ inline v8f wmma_bf16(v16bf a, v16bf b, v8f c) {
    return __builtin_amdgcn_wmma_f32_16x16x32_bf16(
        /*neg_a=*/false, a, /*neg_b=*/false, b,
        /*c_mod=*/(short)0, c, /*reuse_a=*/false, /*reuse_b=*/false);
}

// ---- Tensor Data Mover: 2D bf16 tile load, Global -> LDS ----
// Builds D# group0/group1 per CDNA5 ISA 8.3/8.4; groups 2/3 zero (2D tile).
// This toolchain exposes the 6-arg builtin: (g0, g1, g2, g3, g_extra, cpol).
__device__ inline void tdm_load_2d_bf16(unsigned lds_addr, const void* gptr,
                                        unsigned tensor_d0, unsigned tensor_d1,
                                        unsigned tile_d0, unsigned tile_d1,
                                        unsigned long long stride0_elems) {
    unsigned long long ga = (unsigned long long)(uintptr_t)gptr;
    u32x4 g0;
    g0[0] = 1u;                                        // count=1, user mode
    g0[1] = lds_addr;                                  // LDS byte address
    g0[2] = (unsigned)(ga & 0xffffffffu);              // global_addr[31:0]
    g0[3] = ((unsigned)(ga >> 32) & 0x01ffffffu)       // global_addr[56:32]
            | (2u << 30);                              // type = 2 ("image")
    i32x8 g1;
    g1[0] = (int)(1u << 16);                           // data_size code 1 = 2 bytes
    g1[1] = (int)((tensor_d0 & 0xffffu) << 16);        // tensor_dim0[15:0]
    g1[2] = (int)(((tensor_d0 >> 16) & 0xffffu) |
                  ((tensor_d1 & 0xffffu) << 16));      // dim0 hi | dim1 lo
    g1[3] = (int)(((tensor_d1 >> 16) & 0xffffu) |
                  ((tile_d0 & 0xffffu) << 16));        // dim1 hi | tile_dim0
    g1[4] = (int)(tile_d1 & 0xffffu);                  // tile_dim1, tile_dim2=0
    g1[5] = (int)(unsigned)(stride0_elems & 0xffffffffu);   // dim0_stride[31:0]
    g1[6] = (int)(unsigned)((stride0_elems >> 32) & 0xffffu); // dim0_stride[47:32]
    g1[7] = 0;
    i32x4 z4 = {0, 0, 0, 0};
    i32x8 z8 = {0, 0, 0, 0, 0, 0, 0, 0};
    __builtin_amdgcn_tensor_load_to_lds(g0, g1, z4, z4, z8, 0);
}

// A-fragment (16x32 bf16, MxK): lane -> row = lane&15, kb = lane<16?0:8,
// element i -> k = kb + i + (i>=8?8:0)
__device__ inline v16bf load_a_frag(const __bf16* As, int rowBase, int lane, int lda) {
    const int m  = rowBase + (lane & 15);
    const int kb = (lane < 16) ? 0 : 8;
    v16bf a;
#pragma unroll
    for (int i = 0; i < 16; ++i) {
        const int k = kb + i + ((i >= 8) ? 8 : 0);
        a[i] = As[m * lda + k];
    }
    return a;
}

// B-fragment (32x16 bf16, KxN): lane -> n = base + (lane&15), kb = lane<16?0:8,
// element i -> k = kb + i + (i>=8?8:0)
__device__ inline v16bf load_b_frag(const __bf16* Bs, int ncolBase, int lane, int ldb) {
    v16bf b;
#pragma unroll
    for (int i = 0; i < 16; ++i) b[i] = Bs[(((lane >> 4) * 8) + i + ((i >= 8) ? 8 : 0)) * ldb + ncolBase + (lane & 15)];
    return b;
}

// ---------------- utility kernels ----------------
__global__ __launch_bounds__(256)
void zero_f32_kernel(float* __restrict__ p, long n) {
    long i = (long)blockIdx.x * 256 + threadIdx.x;
    if (i < n) p[i] = 0.0f;
}

// transpose + convert: W [N][K] f32  ->  Wt [K][N] bf16
__global__ __launch_bounds__(256)
void wt_convert_kernel(const float* __restrict__ W, __bf16* __restrict__ Wt, int N, int K) {
    long idx = (long)blockIdx.x * 256 + threadIdx.x;
    if (idx >= (long)N * K) return;
    int n = (int)(idx / K), k = (int)(idx % K);
    Wt[(long)k * N + n] = (__bf16)W[idx];
}

// ---------------- tag scatter (RoPE + gate) ----------------
__global__ __launch_bounds__(256)
void tag_scatter_kernel(const float* __restrict__ we,
                        const int* __restrict__ span_batch,
                        const int* __restrict__ span_start,
                        const int* __restrict__ span_tag,
                        const float* __restrict__ tag_emb,
                        const float* __restrict__ gate_w,
                        const float* __restrict__ gate_b,
                        float* __restrict__ taged) {
    const int sp = blockIdx.x >> 5;     // span index
    const int li = blockIdx.x & 31;     // position within span
    const int b = span_batch[sp];
    const int tag = span_tag[sp];
    const int s = span_start[sp] + li;
    const float* x = we + ((long)(b * SS + s)) * DD;
    const int t = threadIdx.x;

    float partial = 0.f;
#pragma unroll
    for (int pp = 0; pp < 2; ++pp) {
        const int pi = t * 2 + pp;                       // pair index 0..511
        const float inv = __powf(10000.f, -(float)(2 * pi) / (float)DD);
        const float ang = (float)s * inv;
        float sn, cs;
        __sincosf(ang, &sn, &cs);
        const float x1 = x[2 * pi], x2 = x[2 * pi + 1];
        const float r1 = x1 * cs - x2 * sn;
        const float r2 = x1 * sn + x2 * cs;
        partial += r1 * gate_w[2 * pi] + r2 * gate_w[2 * pi + 1];
    }
    __shared__ float red[256];
    red[t] = partial;
    __syncthreads();
    for (int o = 128; o > 0; o >>= 1) {
        if (t < o) red[t] += red[t + o];
        __syncthreads();
    }
    const float dot = red[0] + gate_b[0];
    const float gate = 0.5f / (1.f + __expf(-dot)) + 0.25f;   // GSR=0.5
    const float* te = tag_emb + (long)tag * DD;
    float* outp = taged + ((long)(b * SS + s)) * DD;
#pragma unroll
    for (int pp = 0; pp < 2; ++pp) {
        const int pi = t * 2 + pp;
        atomicAdd(&outp[2 * pi],     gate * te[2 * pi]     * (float)SPANL);
        atomicAdd(&outp[2 * pi + 1], gate * te[2 * pi + 1] * (float)SPANL);
    }
}

// ---------------- layernorm kernels (D = 1024, 256 threads, 4 elem/thread) ----------------
__global__ __launch_bounds__(256)
void ln_tag_kernel(const float* __restrict__ raw, const float* __restrict__ taged,
                   const float* __restrict__ g, const float* __restrict__ be,
                   float* __restrict__ out32, __bf16* __restrict__ outbf) {
    const int row = blockIdx.x, t = threadIdx.x;
    const float* r  = raw   + (long)row * DD;
    const float* tg = taged + (long)row * DD;
    float v[4], s = 0.f, s2 = 0.f;
#pragma unroll
    for (int i = 0; i < 4; ++i) {
        const float x = r[t * 4 + i] + 0.5f * tg[t * 4 + i];  // TAG_RATE=0.5
        v[i] = x; s += x; s2 += x * x;
    }
    __shared__ float rs[256], rq[256];
    rs[t] = s; rq[t] = s2;
    __syncthreads();
    for (int o = 128; o > 0; o >>= 1) {
        if (t < o) { rs[t] += rs[t + o]; rq[t] += rq[t + o]; }
        __syncthreads();
    }
    const float mean = rs[0] * (1.f / DD);
    const float var  = rq[0] * (1.f / DD) - mean * mean;
    const float inv  = rsqrtf(var + 1e-12f);
#pragma unroll
    for (int i = 0; i < 4; ++i) {
        const int c = t * 4 + i;
        const float y = (v[i] - mean) * inv * g[c] + be[c];
        out32[(long)row * DD + c] = y;
        outbf[(long)row * DD + c] = (__bf16)y;
    }
}

__global__ __launch_bounds__(256)
void ln_res_kernel(const float* __restrict__ A, const float* __restrict__ Bb,
                   const float* __restrict__ g, const float* __restrict__ be,
                   float eps, float* __restrict__ out32, __bf16* __restrict__ outbf) {
    const int row = blockIdx.x, t = threadIdx.x;
    const float* a = A  + (long)row * DD;
    const float* bq = Bb + (long)row * DD;
    float v[4], s = 0.f, s2 = 0.f;
#pragma unroll
    for (int i = 0; i < 4; ++i) {
        const float x = a[t * 4 + i] + bq[t * 4 + i];
        v[i] = x; s += x; s2 += x * x;
    }
    __shared__ float rs[256], rq[256];
    rs[t] = s; rq[t] = s2;
    __syncthreads();
    for (int o = 128; o > 0; o >>= 1) {
        if (t < o) { rs[t] += rs[t + o]; rq[t] += rq[t + o]; }
        __syncthreads();
    }
    const float mean = rs[0] * (1.f / DD);
    const float var  = rq[0] * (1.f / DD) - mean * mean;
    const float inv  = rsqrtf(var + eps);
#pragma unroll
    for (int i = 0; i < 4; ++i) {
        const int c = t * 4 + i;
        const float y = (v[i] - mean) * inv * g[c] + be[c];
        out32[(long)row * DD + c] = y;
        outbf[(long)row * DD + c] = (__bf16)y;
    }
}

__global__ __launch_bounds__(256)
void ln_final_kernel(const float* __restrict__ X, const float* __restrict__ FFb,
                     const float* __restrict__ oe,
                     const float* __restrict__ g, const float* __restrict__ be,
                     float eps, float* __restrict__ out) {
    const int row = blockIdx.x, t = threadIdx.x;
    const float* a = X   + (long)row * DD;
    const float* f = FFb + (long)row * DD;
    float v[4], s = 0.f, s2 = 0.f;
#pragma unroll
    for (int i = 0; i < 4; ++i) {
        const float x = a[t * 4 + i] + f[t * 4 + i];
        v[i] = x; s += x; s2 += x * x;
    }
    __shared__ float rs[256], rq[256];
    rs[t] = s; rq[t] = s2;
    __syncthreads();
    for (int o = 128; o > 0; o >>= 1) {
        if (t < o) { rs[t] += rs[t + o]; rq[t] += rq[t + o]; }
        __syncthreads();
    }
    const float mean = rs[0] * (1.f / DD);
    const float var  = rq[0] * (1.f / DD) - mean * mean;
    const float inv  = rsqrtf(var + eps);
#pragma unroll
    for (int i = 0; i < 4; ++i) {
        const int c = t * 4 + i;
        const float y = (v[i] - mean) * inv * g[c] + be[c];
        out[(long)row * DD + c] = y + oe[(long)row * DD + c];
    }
}

// ---------------- WMMA GEMM with TDM tile staging ----------------
// C[M,N] = A[M,K] @ Bt[K,N] + bias
// 256 threads = 8 waves, block tile 128x128, K-step 32.
// Tiles staged Global->LDS by the Tensor Data Mover (wave 0 issues 2 D#'s,
// waits on TENSORcnt, then the workgroup barrier releases the compute waves).
// EPI: 0 = bf16 out, 1 = relu -> bf16 out, 2 = f32 out
template <int EPI>
__global__ __launch_bounds__(256)
void gemm_bf16_kernel(const __bf16* __restrict__ A, const __bf16* __restrict__ Bt,
                      const float* __restrict__ bias, void* __restrict__ Cout,
                      int M, int N, int K) {
    __shared__ alignas(16) __bf16 As[128 * 32];
    __shared__ alignas(16) __bf16 Bs[32 * 128];
    const int m0 = blockIdx.y * 128, n0 = blockIdx.x * 128;
    const int t = threadIdx.x, lane = t & 31, wave = t >> 5;
    const int wm = (wave >> 2) * 64;   // 0 / 64
    const int wn = (wave & 3) * 32;    // 0 / 32 / 64 / 96
    const unsigned ldsA = (unsigned)(uintptr_t)(void*)As;  // addr[31:0] = LDS offset
    const unsigned ldsB = (unsigned)(uintptr_t)(void*)Bs;

    v8f acc[4][2];
#pragma unroll
    for (int mt = 0; mt < 4; ++mt)
#pragma unroll
        for (int nt = 0; nt < 2; ++nt)
#pragma unroll
            for (int i = 0; i < 8; ++i) acc[mt][nt][i] = 0.f;

    for (int k0 = 0; k0 < K; k0 += 32) {
        if (wave == 0) {   // wave-uniform branch: only wave 0 issues the DMAs
            // A tile: 128 rows x 32 cols, row stride K
            tdm_load_2d_bf16(ldsA, A + (long)m0 * K + k0,
                             (unsigned)K, (unsigned)M, 32u, 128u,
                             (unsigned long long)K);
            // B tile: 32 rows x 128 cols, row stride N
            tdm_load_2d_bf16(ldsB, Bt + (long)k0 * N + n0,
                             (unsigned)N, (unsigned)K, 128u, 32u,
                             (unsigned long long)N);
            __builtin_amdgcn_s_wait_tensorcnt(0);
        }
        __syncthreads();

        v16bf bf0 = load_b_frag(Bs, wn, lane, 128);
        v16bf bf1 = load_b_frag(Bs, wn + 16, lane, 128);
#pragma unroll
        for (int mt = 0; mt < 4; ++mt) {
            v16bf af = load_a_frag(As, wm + mt * 16, lane, 32);
            acc[mt][0] = wmma_bf16(af, bf0, acc[mt][0]);
            acc[mt][1] = wmma_bf16(af, bf1, acc[mt][1]);
        }
        __syncthreads();
    }

#pragma unroll
    for (int mt = 0; mt < 4; ++mt)
#pragma unroll
        for (int nt = 0; nt < 2; ++nt) {
            const int n = n0 + wn + nt * 16 + (lane & 15);
            const float bv = bias ? bias[n] : 0.f;
#pragma unroll
            for (int r = 0; r < 8; ++r) {
                const int m = m0 + wm + mt * 16 + r + ((lane >= 16) ? 8 : 0);
                float v = acc[mt][nt][r] + bv;
                if (EPI == 1) v = fmaxf(v, 0.f);
                if (EPI == 2) ((float*)Cout)[(long)m * N + n] = v;
                else          ((__bf16*)Cout)[(long)m * N + n] = (__bf16)v;
            }
        }
}

// ---------------- flash attention (bf16 WMMA, online softmax) ----------------
// grid = (S/64, H, B), 128 threads = 4 waves; wave handles 16 q-rows.
__global__ __launch_bounds__(128)
void attn_kernel(const __bf16* __restrict__ qkv, const int* __restrict__ amask,
                 __bf16* __restrict__ O) {
    __shared__ alignas(16) __bf16 kT[64 * 128];     // [d][j] transposed K chunk
    __shared__ alignas(16) __bf16 Vs[128 * 64];     // [j][d]
    __shared__ float mb[128];
    __shared__ alignas(16) __bf16 Ps[4][16 * 32];   // per-wave P staging

    const int b = blockIdx.z, h = blockIdx.y, q0 = blockIdx.x * 64;
    const int t = threadIdx.x, lane = t & 31, wave = t >> 5;
    const int qr = q0 + wave * 16;

    // load Q fragments (16 rows x 64 d -> two 16x32 A-frags)
    v16bf qf[2];
    {
        const int srow = qr + (lane & 15);
        const __bf16* qp = qkv + ((long)(b * SS + srow)) * (3 * DD) + h * 64;
        const int kb = (lane < 16) ? 0 : 8;
#pragma unroll
        for (int f = 0; f < 2; ++f)
#pragma unroll
            for (int i = 0; i < 16; ++i) {
                const int d = f * 32 + kb + i + ((i >= 8) ? 8 : 0);
                qf[f][i] = qp[d];
            }
    }

    float mrun[8], lrun[8];
    v8f oacc[4];
#pragma unroll
    for (int r = 0; r < 8; ++r) { mrun[r] = -1e30f; lrun[r] = 0.f; }
#pragma unroll
    for (int nt = 0; nt < 4; ++nt)
#pragma unroll
        for (int i = 0; i < 8; ++i) oacc[nt][i] = 0.f;

    for (int jc = 0; jc < SS; jc += 128) {
        { // stage K^T, V and mask bias for this 128-column chunk
            const long rowg = (long)(b * SS + jc + t);
            const __bf16* kp = qkv + rowg * (3 * DD) + DD + h * 64;
            const __bf16* vp = qkv + rowg * (3 * DD) + 2 * DD + h * 64;
#pragma unroll
            for (int d = 0; d < 64; ++d) kT[d * 128 + t] = kp[d];
#pragma unroll
            for (int d = 0; d < 64; ++d) Vs[t * 64 + d] = vp[d];
            mb[t] = (amask[b * SS + jc + t] == 0) ? -1e30f : 0.f;
        }
        __syncthreads();

        for (int jj = 0; jj < 4; ++jj) {   // 32 k-columns per inner step
            v8f sc[2];
#pragma unroll
            for (int ct = 0; ct < 2; ++ct) {
#pragma unroll
                for (int i = 0; i < 8; ++i) sc[ct][i] = 0.f;
                const int colb = jj * 32 + ct * 16;
                v16bf b0, b1;
#pragma unroll
                for (int i = 0; i < 16; ++i) b0[i] = kT[(((lane >> 4) * 8) + i + ((i >= 8) ? 8 : 0)) * 128 + colb + (lane & 15)];
#pragma unroll
                for (int i = 0; i < 16; ++i) b1[i] = kT[(32 + ((lane >> 4) * 8) + i + ((i >= 8) ? 8 : 0)) * 128 + colb + (lane & 15)];
                sc[ct] = wmma_bf16(qf[0], b0, sc[ct]);
                sc[ct] = wmma_bf16(qf[1], b1, sc[ct]);
                const float mbias = mb[colb + (lane & 15)];
#pragma unroll
                for (int r = 0; r < 8; ++r) sc[ct][r] = sc[ct][r] * 0.125f + mbias;
            }

            // online softmax: rowmax / rescale / exp / rowsum (16-lane groups)
            float mnew[8], scl[8];
#pragma unroll
            for (int r = 0; r < 8; ++r) {
                float v = fmaxf(sc[0][r], sc[1][r]);
                for (int off = 8; off >= 1; off >>= 1)
                    v = fmaxf(v, __shfl_xor(v, off, 16));
                mnew[r] = fmaxf(mrun[r], v);
                scl[r] = __expf(mrun[r] - mnew[r]);
                mrun[r] = mnew[r];
            }
#pragma unroll
            for (int ct = 0; ct < 2; ++ct)
#pragma unroll
                for (int r = 0; r < 8; ++r) sc[ct][r] = __expf(sc[ct][r] - mnew[r]);
#pragma unroll
            for (int r = 0; r < 8; ++r) {
                float v = sc[0][r] + sc[1][r];
                for (int off = 8; off >= 1; off >>= 1)
                    v += __shfl_xor(v, off, 16);
                lrun[r] = lrun[r] * scl[r] + v;
            }
#pragma unroll
            for (int nt = 0; nt < 4; ++nt)
#pragma unroll
                for (int r = 0; r < 8; ++r) oacc[nt][r] *= scl[r];

            // stage P (16x32) into per-wave LDS in C-layout, reload as A-frag
#pragma unroll
            for (int ct = 0; ct < 2; ++ct)
#pragma unroll
                for (int r = 0; r < 8; ++r) {
                    const int row = r + ((lane >= 16) ? 8 : 0);
                    Ps[wave][row * 32 + ct * 16 + (lane & 15)] = (__bf16)sc[ct][r];
                }
            v16bf pf;
            {
                const int kb = (lane < 16) ? 0 : 8;
                const int row = lane & 15;
#pragma unroll
                for (int i = 0; i < 16; ++i) {
                    const int k = kb + i + ((i >= 8) ? 8 : 0);
                    pf[i] = Ps[wave][row * 32 + k];
                }
            }
            // O += P (16x32) @ V (32x64)
#pragma unroll
            for (int nt = 0; nt < 4; ++nt) {
                v16bf vf;
#pragma unroll
                for (int i = 0; i < 16; ++i)
                    vf[i] = Vs[(jj * 32 + ((lane >> 4) * 8) + i + ((i >= 8) ? 8 : 0)) * 64 + nt * 16 + (lane & 15)];
                oacc[nt] = wmma_bf16(pf, vf, oacc[nt]);
            }
        }
        __syncthreads();
    }

    // normalize and write O (bf16, [B*S][D] with head offset)
#pragma unroll
    for (int nt = 0; nt < 4; ++nt)
#pragma unroll
        for (int r = 0; r < 8; ++r) {
            const int row = qr + r + ((lane >= 16) ? 8 : 0);
            const int col = h * 64 + nt * 16 + (lane & 15);
            O[(long)(b * SS + row) * DD + col] = (__bf16)(oacc[nt][r] / lrun[r]);
        }
}

// ---------------- host launcher ----------------
extern "C" void kernel_launch(void* const* d_in, const int* in_sizes, int n_in,
                              void* d_out, int out_size, void* d_ws, size_t ws_size,
                              hipStream_t stream) {
    const float* we         = (const float*)d_in[0];
    const int*   amask      = (const int*)d_in[1];
    const int*   span_batch = (const int*)d_in[2];
    const int*   span_start = (const int*)d_in[3];
    const int*   span_tag   = (const int*)d_in[4];
    const float* tag_emb    = (const float*)d_in[5];
    const float* gate_w     = (const float*)d_in[6];
    const float* gate_b     = (const float*)d_in[7];
    const float* attn_ln_g  = (const float*)d_in[8];
    const float* attn_ln_b  = (const float*)d_in[9];
    const float* in_proj_w  = (const float*)d_in[10];
    const float* in_proj_b  = (const float*)d_in[11];
    const float* out_proj_w = (const float*)d_in[12];
    const float* out_proj_b = (const float*)d_in[13];
    const float* enc_ln1_g  = (const float*)d_in[14];
    const float* enc_ln1_b  = (const float*)d_in[15];
    const float* lin1_w     = (const float*)d_in[16];
    const float* lin1_b     = (const float*)d_in[17];
    const float* lin2_w     = (const float*)d_in[18];
    const float* lin2_b     = (const float*)d_in[19];
    const float* enc_ln2_g  = (const float*)d_in[20];
    const float* enc_ln2_b  = (const float*)d_in[21];
    float* outp = (float*)d_out;

    const long TOK = (long)BB * SS;          // 8192 tokens
    char* ws = (char*)d_ws;
    size_t off = 0;
    auto alloc = [&](size_t bytes) -> void* {
        void* p = ws + off;
        off = (off + bytes + 255) & ~(size_t)255;
        return p;
    };

    float*  taged   = (float*) alloc(TOK * DD * sizeof(float));
    float*  out_emb = (float*) alloc(TOK * DD * sizeof(float));
    __bf16* xe      = (__bf16*)alloc(TOK * DD * sizeof(__bf16));
    __bf16* wqkvT   = (__bf16*)alloc((size_t)DD * 3 * DD * sizeof(__bf16));
    __bf16* woutT   = (__bf16*)alloc((size_t)DD * DD * sizeof(__bf16));
    __bf16* w1T     = (__bf16*)alloc((size_t)DD * FF * sizeof(__bf16));
    __bf16* w2T     = (__bf16*)alloc((size_t)FF * DD * sizeof(__bf16));
    __bf16* qkvb    = (__bf16*)alloc(TOK * 3 * DD * sizeof(__bf16));
    __bf16* Ob      = (__bf16*)alloc(TOK * DD * sizeof(__bf16));
    float*  attnf   = (float*) alloc(TOK * DD * sizeof(float));
    float*  xf      = (float*) alloc(TOK * DD * sizeof(float));
    __bf16* xbf     = (__bf16*)alloc(TOK * DD * sizeof(__bf16));
    __bf16* hbuf    = (__bf16*)alloc(TOK * FF * sizeof(__bf16));
    float*  ffb     = (float*) alloc(TOK * DD * sizeof(float));
    (void)ws_size; (void)in_sizes; (void)n_in; (void)out_size;

    // 1. zero taged, scatter gated tags
    {
        long n = TOK * DD;
        zero_f32_kernel<<<(unsigned)((n + 255) / 256), 256, 0, stream>>>(taged, n);
        tag_scatter_kernel<<<NSPAN * SPANL, 256, 0, stream>>>(
            we, span_batch, span_start, span_tag, tag_emb, gate_w, gate_b, taged);
    }
    // 2. out_emb = LN(raw + 0.5*taged), eps=1e-12; also bf16 copy
    ln_tag_kernel<<<(unsigned)TOK, 256, 0, stream>>>(we, taged, attn_ln_g, attn_ln_b,
                                                     out_emb, xe);
    // 3. weight transpose+convert
    {
        long n;
        n = (long)3 * DD * DD;
        wt_convert_kernel<<<(unsigned)((n + 255) / 256), 256, 0, stream>>>(in_proj_w, wqkvT, 3 * DD, DD);
        n = (long)DD * DD;
        wt_convert_kernel<<<(unsigned)((n + 255) / 256), 256, 0, stream>>>(out_proj_w, woutT, DD, DD);
        n = (long)FF * DD;
        wt_convert_kernel<<<(unsigned)((n + 255) / 256), 256, 0, stream>>>(lin1_w, w1T, FF, DD);
        n = (long)DD * FF;
        wt_convert_kernel<<<(unsigned)((n + 255) / 256), 256, 0, stream>>>(lin2_w, w2T, DD, FF);
    }
    // 4. QKV = xe @ wqkvT + b    (8192 x 3072 x 1024)
    gemm_bf16_kernel<0><<<dim3(3 * DD / 128, (unsigned)(TOK / 128)), 256, 0, stream>>>(
        xe, wqkvT, in_proj_b, qkvb, (int)TOK, 3 * DD, DD);
    // 5. flash attention -> Ob
    attn_kernel<<<dim3(SS / 64, HH, BB), 128, 0, stream>>>(qkvb, amask, Ob);
    // 6. attn proj = Ob @ woutT + b  (f32 out)
    gemm_bf16_kernel<2><<<dim3(DD / 128, (unsigned)(TOK / 128)), 256, 0, stream>>>(
        Ob, woutT, out_proj_b, attnf, (int)TOK, DD, DD);
    // 7. x = LN(out_emb + attn), eps=1e-5
    ln_res_kernel<<<(unsigned)TOK, 256, 0, stream>>>(out_emb, attnf, enc_ln1_g, enc_ln1_b,
                                                     1e-5f, xf, xbf);
    // 8. h = relu(x @ w1T + b1)  (8192 x 4096 x 1024, bf16 out)
    gemm_bf16_kernel<1><<<dim3(FF / 128, (unsigned)(TOK / 128)), 256, 0, stream>>>(
        xbf, w1T, lin1_b, hbuf, (int)TOK, FF, DD);
    // 9. ff = h @ w2T + b2       (8192 x 1024 x 4096, f32 out)
    gemm_bf16_kernel<2><<<dim3(DD / 128, (unsigned)(TOK / 128)), 256, 0, stream>>>(
        hbuf, w2T, lin2_b, ffb, (int)TOK, DD, FF);
    // 10. out = LN(x + ff)*g+b + out_emb, eps=1e-5
    ln_final_kernel<<<(unsigned)TOK, 256, 0, stream>>>(xf, ffb, out_emb, enc_ln2_g, enc_ln2_b,
                                                       1e-5f, outp);
}
